// MyAttentionHead_24172075942194
// MI455X (gfx1250) — matrix operand order repaired and checked by reading, not compiled, not run on hardware
//
#include <hip/hip_runtime.h>
#include <hip/hip_bf16.h>

// ---------------------------------------------------------------------------
// Single-head causal attention for MI455X (gfx1250, wave32, WMMA).
//   x:[4,2048,1024] f32, Wq/Wk/Wv:[64,1024] f32  ->  out:[4,2048,64] f32
// Pass 0: one-shot f32->bf16 weight conversion into ws (L2-resident, 384 KB)
// Pass 1: bf16 WMMA projection, software-pipelined (all 12 B-frags loaded
//         into distinct regs per K-step; next A-frag prefetched).
// Pass 2: flash attention, 16-query tile per wave, 32-key blocks; V loads
//         hoisted above softmax so their latency hides under VALU work.
// ---------------------------------------------------------------------------

#define D_MODEL 1024
#define D_HEAD  64
#define BATCH   4
#define SEQ     2048
#define NTOK    (BATCH * SEQ)          // 8192 tokens
#define SCALE   0.03125f               // 1/sqrt(1024)

typedef __bf16 v16bf __attribute__((ext_vector_type(16)));
typedef float  v8f   __attribute__((ext_vector_type(8)));

union BFrag {                 // 16 bf16 = 32 bytes = 8 VGPRs per lane
    v16bf v;
    __bf16 h[16];
    unsigned short u[16];
    uint4 q[2];
};

// Hardware f32->bf16 (v_cvt_pk_bf16_f32 on gfx1250), result as raw bits.
__device__ __forceinline__ unsigned short f2bf(float f) {
    __bf16 h = (__bf16)f;
    return __builtin_bit_cast(unsigned short, h);
}

__device__ __forceinline__ void pack8(__bf16* d, float4 a, float4 b) {
    d[0] = (__bf16)a.x; d[1] = (__bf16)a.y; d[2] = (__bf16)a.z; d[3] = (__bf16)a.w;
    d[4] = (__bf16)b.x; d[5] = (__bf16)b.y; d[6] = (__bf16)b.z; d[7] = (__bf16)b.w;
}

__device__ __forceinline__ v8f wmma_bf16(const BFrag& a, const BFrag& b, v8f c) {
    return __builtin_amdgcn_wmma_f32_16x16x32_bf16(
        /*neg_a=*/false, a.v, /*neg_b=*/false, b.v,
        /*c_mod=*/(short)0, c, /*reuse_a=*/false, /*reuse_b=*/false);
}

// ---------------------------------------------------------------------------
// Kernel 0: convert the three weight matrices to bf16 once.
//   Wb layout: [3][64][1024] bf16 row-major, m stride 65536.
// ---------------------------------------------------------------------------
__global__ __launch_bounds__(256) void wconv_kernel(
    const float* __restrict__ Wq, const float* __restrict__ Wk,
    const float* __restrict__ Wv, unsigned short* __restrict__ Wb)
{
    const int i = (blockIdx.x * 256 + threadIdx.x) * 4;   // 4 elems/thread
    const int n = D_HEAD * D_MODEL;                       // 65536 per matrix
    const float* src = (i < n) ? Wq : (i < 2 * n) ? Wk : Wv;
    const int off = i & (n - 1);
    float4 f = *reinterpret_cast<const float4*>(src + off);
    ushort4 o;
    o.x = f2bf(f.x); o.y = f2bf(f.y); o.z = f2bf(f.z); o.w = f2bf(f.w);
    *reinterpret_cast<ushort4*>(Wb + i) = o;
}

// ---------------------------------------------------------------------------
// Kernel 1: q/k/v projection.  One wave per 16-token tile; 12 WMMA / k-step.
// All 12 B-fragments get distinct registers so loads issue as one batch and
// WMMAs drain them with progressive waits; next A-frag is prefetched.
// ---------------------------------------------------------------------------
__global__ __launch_bounds__(128) void qkv_proj_kernel(
    const float* __restrict__ x,
    const unsigned short* __restrict__ Wb,   // [3][64][1024] bf16
    unsigned short* __restrict__ Qb,         // [NTOK][64] bf16
    unsigned short* __restrict__ Kb,         // [NTOK][64] bf16
    unsigned short* __restrict__ Vt)         // [BATCH][64][SEQ] bf16 (transposed)
{
    const int lane = threadIdx.x & 31;
    const int wave = threadIdx.x >> 5;
    const int tile = blockIdx.x * 4 + wave;          // 0..511
    const int t0   = tile * 16;
    const int lx   = lane & 15;
    const int kh   = (lane >= 16) ? 8 : 0;           // unused (A-frag also uses kb16)
    const int kb16 = (lane >= 16) ? 16 : 0;          // B-frag K-half offset

    v8f acc[12];
#pragma unroll
    for (int i = 0; i < 12; ++i) acc[i] = (v8f){};

    const float* xrow = x + (long)(t0 + lx) * D_MODEL;
    // Per-lane bf16 weight base; fragment i (= m*4+nt) lives at i*16384 + k0.
    const unsigned short* wlane = Wb + lx * D_MODEL + kb16;

    // Preload A-fragment raw f32 for k0 = 0.
    const float4* xp = reinterpret_cast<const float4*>(xrow + kb16);
    float4 a0 = xp[0], a1 = xp[1], a2 = xp[2], a3 = xp[3];

    for (int k0 = 0; k0 < D_MODEL; k0 += 32) {
        // ---- current A fragment (converted from prefetched f32) ----
        BFrag a;
        pack8(a.h + 0, a0, a1);                      // k = k0+kb16+0..7
        pack8(a.h + 8, a2, a3);                      // k = k0+kb16+8..15

        // ---- issue ALL 24 B-fragment loads (distinct registers) ----
        BFrag bfr[12];
#pragma unroll
        for (int i = 0; i < 12; ++i) {
            const unsigned short* wp = wlane + (long)i * 16384 + k0;
            bfr[i].q[0] = *reinterpret_cast<const uint4*>(wp + 0);
            bfr[i].q[1] = *reinterpret_cast<const uint4*>(wp + 8);
        }

        // ---- prefetch next A-fragment raw f32 (clamped, branch-free) ----
        const int kn = (k0 + 32 < D_MODEL) ? k0 + 32 : k0;
        const float4* xpn = reinterpret_cast<const float4*>(xrow + kn + kb16);
        a0 = xpn[0]; a1 = xpn[1]; a2 = xpn[2]; a3 = xpn[3];

        // ---- 12 WMMAs drain the loads with progressive waits ----
#pragma unroll
        for (int i = 0; i < 12; ++i)
            acc[i] = wmma_bf16(a, bfr[i], acc[i]);
    }

    // ---- store bf16 results (Q,K row-major; V transposed) ----
    const int rbase = (lane >= 16) ? 8 : 0;
#pragma unroll
    for (int m = 0; m < 3; ++m) {
#pragma unroll
        for (int nt = 0; nt < 4; ++nt) {
#pragma unroll
            for (int r = 0; r < 8; ++r) {
                const int tok = t0 + r + rbase;
                const int d   = nt * 16 + lx;
                const unsigned short h = f2bf(acc[m * 4 + nt][r]);
                if (m == 0)      Qb[(long)tok * D_HEAD + d] = h;
                else if (m == 1) Kb[(long)tok * D_HEAD + d] = h;
                else {
                    const int b = tok >> 11, s = tok & (SEQ - 1);
                    Vt[((long)b * D_HEAD + d) * SEQ + s] = h;
                }
            }
        }
    }
}

// ---------------------------------------------------------------------------
// Kernel 2: flash attention.  One wave per 16-query tile, 32-key blocks.
// K and V fragment loads are both issued at the top of each iteration; V's
// latency hides under the QK WMMAs + softmax VALU stretch.
// ---------------------------------------------------------------------------
__global__ __launch_bounds__(128) void flash_attn_kernel(
    const unsigned short* __restrict__ Qb,
    const unsigned short* __restrict__ Kb,
    const unsigned short* __restrict__ Vt,
    float* __restrict__ out)
{
    __shared__ unsigned short plds[4][16 * 32];      // P staging per wave

    const int lane = threadIdx.x & 31;
    const int wave = threadIdx.x >> 5;
    const int tile = blockIdx.x * 4 + wave;          // 0..511
    const int b    = tile >> 7;                      // batch
    const int qt   = tile & 127;                     // query tile in batch
    const int lx   = lane & 15;
    const int kh   = (lane >= 16) ? 8 : 0;
    const int kb16 = (lane >= 16) ? 16 : 0;
    const int rbase = (lane >= 16) ? 8 : 0;

    // ---- load Q tile as two A fragments (d_head = 2 x 32) ----
    const unsigned short* qrow = Qb + ((long)b * SEQ + qt * 16 + lx) * D_HEAD;
    BFrag qf[2];
#pragma unroll
    for (int c = 0; c < 2; ++c) {
        qf[c].q[0] = *reinterpret_cast<const uint4*>(qrow + c * 32 + kb16);
        qf[c].q[1] = *reinterpret_cast<const uint4*>(qrow + c * 32 + kb16 + 8);
    }

    v8f o[4];
#pragma unroll
    for (int nt = 0; nt < 4; ++nt) o[nt] = (v8f){};
    float ms[8], ls[8];
#pragma unroll
    for (int r = 0; r < 8; ++r) { ms[r] = -3.0e38f; ls[r] = 0.0f; }

    const int jmax = (qt * 16 + 15) >> 5;            // causal block limit

    for (int jb = 0; jb <= jmax; ++jb) {
        const int kbase = jb * 32;

        // ---- issue K-fragment AND V-fragment loads up front ----
        BFrag kf[2][2];                              // [ntile][k-chunk]
#pragma unroll
        for (int nt = 0; nt < 2; ++nt) {
            const unsigned short* krow =
                Kb + ((long)b * SEQ + kbase + nt * 16 + lx) * D_HEAD + kb16;
            kf[nt][0].q[0] = *reinterpret_cast<const uint4*>(krow + 0);
            kf[nt][0].q[1] = *reinterpret_cast<const uint4*>(krow + 8);
            kf[nt][1].q[0] = *reinterpret_cast<const uint4*>(krow + 32);
            kf[nt][1].q[1] = *reinterpret_cast<const uint4*>(krow + 40);
        }
        BFrag vf[4];                                 // V^T: contiguous in key
#pragma unroll
        for (int nt = 0; nt < 4; ++nt) {
            const unsigned short* vp =
                Vt + ((long)b * D_HEAD + nt * 16 + lx) * SEQ + kbase + kb16;
            vf[nt].q[0] = *reinterpret_cast<const uint4*>(vp + 0);
            vf[nt].q[1] = *reinterpret_cast<const uint4*>(vp + 8);
        }

        // ---- S = Q * K^T  (two 16-key n-tiles, K=64 as 2x32) ----
        v8f sc[2];
#pragma unroll
        for (int nt = 0; nt < 2; ++nt) {
            sc[nt] = wmma_bf16(qf[0], kf[nt][0], (v8f){});
            sc[nt] = wmma_bf16(qf[1], kf[nt][1], sc[nt]);
        }

        // ---- scale + causal mask ----
        float p[2][8];
#pragma unroll
        for (int nt = 0; nt < 2; ++nt)
#pragma unroll
            for (int r = 0; r < 8; ++r) {
                const int row = qt * 16 + r + rbase;
                const int col = kbase + nt * 16 + lx;
                float v = sc[nt][r] * SCALE;
                p[nt][r] = (col > row) ? -3.0e38f : v;
            }

        // ---- online softmax (row reductions across 16-lane halves) ----
#pragma unroll
        for (int r = 0; r < 8; ++r) {
            float mr = fmaxf(p[0][r], p[1][r]);
            mr = fmaxf(mr, __shfl_xor(mr, 1, 32));
            mr = fmaxf(mr, __shfl_xor(mr, 2, 32));
            mr = fmaxf(mr, __shfl_xor(mr, 4, 32));
            mr = fmaxf(mr, __shfl_xor(mr, 8, 32));
            const float mnew  = fmaxf(ms[r], mr);
            const float alpha = __expf(ms[r] - mnew);
            const float p0 = __expf(p[0][r] - mnew);
            const float p1 = __expf(p[1][r] - mnew);
            float rs = p0 + p1;
            rs += __shfl_xor(rs, 1, 32);
            rs += __shfl_xor(rs, 2, 32);
            rs += __shfl_xor(rs, 4, 32);
            rs += __shfl_xor(rs, 8, 32);
            ls[r] = ls[r] * alpha + rs;
            ms[r] = mnew;
            p[0][r] = p0;
            p[1][r] = p1;
#pragma unroll
            for (int nt = 0; nt < 4; ++nt) o[nt][r] *= alpha;
        }

        // ---- C-layout -> A-layout reshape of P through LDS ----
        unsigned short* L = plds[wave];
#pragma unroll
        for (int nt = 0; nt < 2; ++nt)
#pragma unroll
            for (int r = 0; r < 8; ++r)
                L[(r + rbase) * 32 + nt * 16 + lx] = f2bf(p[nt][r]);
        asm volatile("s_wait_dscnt 0" ::: "memory");

        BFrag pf;
        pf.q[0] = *reinterpret_cast<const uint4*>(&L[lx * 32 + kb16]);
        pf.q[1] = *reinterpret_cast<const uint4*>(&L[lx * 32 + kb16 + 8]);
        asm volatile("s_wait_dscnt 0" ::: "memory");

        // ---- O += P * V  (V fragments already resident) ----
#pragma unroll
        for (int nt = 0; nt < 4; ++nt)
            o[nt] = wmma_bf16(pf, vf[nt], o[nt]);
    }

    // ---- normalize and store f32 output ----
#pragma unroll
    for (int nt = 0; nt < 4; ++nt)
#pragma unroll
        for (int r = 0; r < 8; ++r) {
            const int row = qt * 16 + r + rbase;
            const int d   = nt * 16 + lx;
            out[((long)b * SEQ + row) * D_HEAD + d] = o[nt][r] / ls[r];
        }
}

// ---------------------------------------------------------------------------
extern "C" void kernel_launch(void* const* d_in, const int* in_sizes, int n_in,
                              void* d_out, int out_size, void* d_ws, size_t ws_size,
                              hipStream_t stream) {
    const float* x  = (const float*)d_in[0];
    const float* Wq = (const float*)d_in[1];
    const float* Wk = (const float*)d_in[2];
    const float* Wv = (const float*)d_in[3];

    unsigned short* Qb = (unsigned short*)d_ws;               // 1 MB
    unsigned short* Kb = Qb + (size_t)NTOK * D_HEAD;          // 1 MB
    unsigned short* Vt = Kb + (size_t)NTOK * D_HEAD;          // 1 MB
    unsigned short* Wb = Vt + (size_t)NTOK * D_HEAD;          // 384 KB bf16 weights

    // Pass 0: 3*64*1024 = 196608 f32 -> bf16, 4 per thread.
    wconv_kernel<<<192, 256, 0, stream>>>(Wq, Wk, Wv, Wb);
    // 512 sixteen-token tiles, 4 waves (128 threads) per block.
    qkv_proj_kernel<<<128, 128, 0, stream>>>(x, Wb, Qb, Kb, Vt);
    flash_attn_kernel<<<128, 128, 0, stream>>>(Qb, Kb, Vt, (float*)d_out);
}
